// MultiModalFusionNetwork_36069135352338
// MI455X (gfx1250) — matrix operand order repaired and checked by reading, not compiled, not run on hardware
//
#include <hip/hip_runtime.h>
#include <hip/hip_bf16.h>

// ---------------------------------------------------------------------------
// MultiModalFusionNetwork on gfx1250 (MI455X).
// All GEMMs: v_wmma_f32_16x16x32_bf16, weights pre-packed to fragment layout,
// compile-time K -> fully unrolled software-pipelined K loop (no mid-loop
// exits, accumulator coalesced in place), fused bias/LN/residual/GELU
// epilogues.
// ---------------------------------------------------------------------------

typedef __attribute__((ext_vector_type(16))) __bf16 v16bf;
typedef __attribute__((ext_vector_type(8)))  float  v8f;

#define BROWS 16384   // batch
#define LDIM  256     // hidden width

enum { PRE_NONE = 0, PRE_GELU = 1 };
enum { EPI_LN = 0, EPI_LN_REG = 1, EPI_PLAIN = 2, EPI_ATTN_FIRST = 3,
       EPI_ATTN = 4, EPI_LN_GELU = 5 };

__device__ __forceinline__ float gelu_exact(float x) {
  return 0.5f * x * (1.0f + erff(x * 0.70710678118654752f));
}
__device__ __forceinline__ void gelu4(float4& v) {
  v.x = gelu_exact(v.x); v.y = gelu_exact(v.y);
  v.z = gelu_exact(v.z); v.w = gelu_exact(v.w);
}
__device__ __forceinline__ v16bf pack_a(float4 q0, float4 q1,
                                        float4 q2, float4 q3) {
  v16bf a;
  a[0]=(__bf16)q0.x;  a[1]=(__bf16)q0.y;  a[2]=(__bf16)q0.z;  a[3]=(__bf16)q0.w;
  a[4]=(__bf16)q1.x;  a[5]=(__bf16)q1.y;  a[6]=(__bf16)q1.z;  a[7]=(__bf16)q1.w;
  a[8]=(__bf16)q2.x;  a[9]=(__bf16)q2.y;  a[10]=(__bf16)q2.z; a[11]=(__bf16)q2.w;
  a[12]=(__bf16)q3.x; a[13]=(__bf16)q3.y; a[14]=(__bf16)q3.z; a[15]=(__bf16)q3.w;
  return a;
}

struct GArgs {
  const float*  A;        // B x K (row-major, stride lda); unused when CAT5
  const float*  cat[5];   // virtual concat sources (CAT5 mode), each B x 256
  const __bf16* PW;       // packed weights in WMMA B-fragment layout
  const float*  bias;     // N
  const float*  g;        // LN gamma (N)
  const float*  bvec;     // LN beta  (N)
  const float*  add_in;   // residual input (B x N) for ATTN epilogues
  const float*  regime;   // 6 x 256
  const int*    ridx;     // B regime indices
  float*        out;      // B x N output (accumulator for EPI_ATTN)
  int lda;                // A row stride
};

// ---------------------------------------------------------------------------
// Weight packing: fp32 KxN (row-major) -> bf16 fragments.
// Fragment (s, t, lane) holds 16 contiguous bf16: element e corresponds to
// K = s*32 + (lane>=16)*16 + e, Ncol = t*16 + (lane&15).  (contiguous k order, same as the A operand)
// ---------------------------------------------------------------------------

__global__ __launch_bounds__(256)
void pack_w(const float* __restrict__ src, __bf16* __restrict__ dst,
            int Kp, int N, int Kreal) {
  int NT = N >> 4;
  int nfrag = (Kp >> 5) * NT * 32;
  int fid = blockIdx.x * 256 + threadIdx.x;
  if (fid >= nfrag) return;
  int l = fid & 31;
  int t = (fid >> 5) % NT;
  int s = fid / (32 * NT);
  int kh = l >> 4, nn = t * 16 + (l & 15);
  v16bf v;
  #pragma unroll
  for (int e = 0; e < 16; ++e) {
    int kk = s * 32 + kh * 16 + e;
    float f = (kk < Kreal) ? src[(size_t)kk * N + nn] : 0.0f;
    v[e] = (__bf16)f;
  }
  *(v16bf*)(dst + (size_t)fid * 16) = v;
}

struct PackB { const float* src[125]; };  // all the 256x256 weights

__global__ __launch_bounds__(256)
void pack_w256(PackB pb, __bf16* __restrict__ dst) {
  int widx = blockIdx.y;
  int fid  = blockIdx.x * 256 + threadIdx.x;        // 0..4095
  const float* __restrict__ src = pb.src[widx];
  int l = fid & 31, t = (fid >> 5) & 15, s = fid >> 9;
  int kh = l >> 4, nn = t * 16 + (l & 15);
  v16bf v;
  #pragma unroll
  for (int e = 0; e < 16; ++e) {
    int kk = s * 32 + kh * 16 + e;
    v[e] = (__bf16)src[(size_t)kk * 256 + nn];
  }
  *(v16bf*)(dst + (size_t)widx * 65536 + (size_t)fid * 16) = v;
}

// Zero-pad B x d input to B x 32.
__global__ __launch_bounds__(256)
void pad_x(const float* __restrict__ x, float* __restrict__ out, int d) {
  int i = blockIdx.x * 256 + threadIdx.x;   // over B*32
  int r = i >> 5, c = i & 31;
  out[i] = (c < d) ? x[r * d + c] : 0.0f;
}

// ---------------------------------------------------------------------------
// GEMM + fused epilogue.  Workgroup = 256 threads (8 wave32), computes rows
// [16*blockIdx.x, +16) x N.  Each wave owns N/128 16x16 WMMA tiles.
// KC is compile-time: the pipelined loop fully unrolls, all pipeline guards
// constant-fold, accumulators stay in place.
// ---------------------------------------------------------------------------

template <int N, int KC, int PRE, int EPI, bool CAT5>
__global__ __launch_bounds__(256)
void gemm_epi_kernel(GArgs ga) {
  constexpr int WPT    = N / 128;       // 16x16 tiles per wave (1 or 2)
  constexpr int NT     = N / 16;        // n-tiles total
  constexpr int NSTEPS = KC / 32;
  constexpr int NPAIR  = (NSTEPS + 1) / 2;
  __shared__ float tile[16][N];
  __shared__ float psum[16][16];
  __shared__ float psq [16][16];

  const int tid  = threadIdx.x;
  const int wave = tid >> 5;
  const int lane = tid & 31;
  const int row0 = blockIdx.x * 16;
  const int mA   = lane & 15;
  const int kh   = lane >> 4;

  const int lda = CAT5 ? 256 : ga.lda;
  const size_t rowoff = (size_t)(row0 + mA) * 256 + kh * 16;  // CAT5 row offset

  const float*  ap = (CAT5 ? ga.cat[0] : ga.A)
                     + (size_t)(row0 + mA) * lda + kh * 16;
  const __bf16* bp = ga.PW + ((size_t)(wave * WPT) * 512 + (size_t)lane * 16);

  v8f c0 = {}, c1 = {};

  float4 p00, p01, p02, p03, p10, p11, p12, p13;
  v16bf  f00 = {}, f01 = {}, f10 = {}, f11 = {};

  // prologue: step 0 into buffer 0
  p00 = ((const float4*)ap)[0];
  p01 = ((const float4*)ap)[1];
  p02 = ((const float4*)(ap + 8))[0];
  p03 = ((const float4*)(ap + 8))[1];
  f00 = *(const v16bf*)bp;
  if (WPT == 2) f01 = *(const v16bf*)(bp + 512);

  int snext = 1;

#define PREFETCH(P0, P1, P2, P3, F0, F1)                                  \
  do {                                                                    \
    if (snext < NSTEPS) {                                                 \
      if (CAT5 && ((snext & 7) == 0)) ap = ga.cat[snext >> 3] + rowoff;   \
      else                            ap += 32;                           \
      bp += NT * 512;                                                     \
      P0 = ((const float4*)ap)[0];                                        \
      P1 = ((const float4*)ap)[1];                                        \
      P2 = ((const float4*)(ap + 8))[0];                                  \
      P3 = ((const float4*)(ap + 8))[1];                                  \
      F0 = *(const v16bf*)bp;                                             \
      if (WPT == 2) F1 = *(const v16bf*)(bp + 512);                       \
    }                                                                     \
    ++snext;                                                              \
  } while (0)

#define COMPUTE(P0, P1, P2, P3, F0, F1)                                   \
  do {                                                                    \
    float4 t0 = P0, t1 = P1, t2 = P2, t3 = P3;                            \
    if (PRE == PRE_GELU) { gelu4(t0); gelu4(t1); gelu4(t2); gelu4(t3); }  \
    v16bf a = pack_a(t0, t1, t2, t3);                                     \
    c0 = __builtin_amdgcn_wmma_f32_16x16x32_bf16(                         \
             false, a, false, F0, (short)0, c0, false, false);            \
    if (WPT == 2)                                                         \
      c1 = __builtin_amdgcn_wmma_f32_16x16x32_bf16(                       \
               false, a, false, F1, (short)0, c1, false, false);          \
  } while (0)

  #pragma unroll
  for (int it = 0; it < NPAIR; ++it) {
    PREFETCH(p10, p11, p12, p13, f10, f11);
    COMPUTE (p00, p01, p02, p03, f00, f01);
    if (2 * it + 1 < NSTEPS) {            // constant after full unroll
      PREFETCH(p00, p01, p02, p03, f00, f01);
      COMPUTE (p10, p11, p12, p13, f10, f11);
    }
  }
#undef PREFETCH
#undef COMPUTE

  // Stage C (+bias) into LDS: VGPR r -> row r + 8*(lane>=16).
  {
    const int rbase = (lane >> 4) * 8;
    int ncol = wave * (16 * WPT) + (lane & 15);
    float bb = ga.bias[ncol];
    #pragma unroll
    for (int r = 0; r < 8; ++r) tile[rbase + r][ncol] = c0[r] + bb;
    if (WPT == 2) {
      ncol += 16;
      bb = ga.bias[ncol];
      #pragma unroll
      for (int r = 0; r < 8; ++r) tile[rbase + r][ncol] = c1[r] + bb;
    }
  }
  __syncthreads();

  // Epilogue: 16 threads per row, each owns N/16 contiguous columns.
  const int row = tid >> 4;
  const int seg = tid & 15;
  constexpr int CH = N / 16;
  const int c0i  = seg * CH;
  const int grow = row0 + row;
  const size_t ob = (size_t)grow * N + c0i;

  if (EPI == EPI_PLAIN) {
    #pragma unroll
    for (int i = 0; i < CH; i += 4)
      *(float4*)&ga.out[ob + i] = *(float4*)&tile[row][c0i + i];
    return;
  }

  if (EPI == EPI_ATTN || EPI == EPI_ATTN_FIRST) {
    #pragma unroll
    for (int i = 0; i < CH; i += 4) {
      float4 av = *(const float4*)&ga.add_in[ob + i];
      tile[row][c0i + i + 0] += av.x;
      tile[row][c0i + i + 1] += av.y;
      tile[row][c0i + i + 2] += av.z;
      tile[row][c0i + i + 3] += av.w;
    }
  }

  float s1 = 0.0f, sq = 0.0f;
  #pragma unroll
  for (int i = 0; i < CH; ++i) {
    float x = tile[row][c0i + i];
    s1 += x; sq += x * x;
  }
  psum[row][seg] = s1;
  psq [row][seg] = sq;
  __syncthreads();

  float S = 0.0f, SQ = 0.0f;
  #pragma unroll
  for (int i = 0; i < 16; ++i) { S += psum[row][i]; SQ += psq[row][i]; }
  float mean = S * (1.0f / N);
  float var  = SQ * (1.0f / N) - mean * mean;
  float rstd = rsqrtf(var + 1e-5f);

  int rix = 0;
  if (EPI == EPI_LN_REG) rix = ga.ridx[grow];

  #pragma unroll
  for (int i = 0; i < CH; i += 4) {
    float4 gg = *(const float4*)&ga.g[c0i + i];
    float4 bb = *(const float4*)&ga.bvec[c0i + i];
    float4 xv = *(float4*)&tile[row][c0i + i];
    float4 y;
    y.x = (xv.x - mean) * rstd * gg.x + bb.x;
    y.y = (xv.y - mean) * rstd * gg.y + bb.y;
    y.z = (xv.z - mean) * rstd * gg.z + bb.z;
    y.w = (xv.w - mean) * rstd * gg.w + bb.w;
    if (EPI == EPI_LN_REG) {
      float4 rg = *(const float4*)&ga.regime[rix * N + c0i + i];
      y.x += rg.x; y.y += rg.y; y.z += rg.z; y.w += rg.w;
    } else if (EPI == EPI_LN_GELU) {
      gelu4(y);
    } else if (EPI == EPI_ATTN_FIRST) {
      float4 av = *(const float4*)&ga.add_in[ob + i];
      y.x += av.x; y.y += av.y; y.z += av.z; y.w += av.w;
    } else if (EPI == EPI_ATTN) {
      float4 ov = *(float4*)&ga.out[ob + i];
      y.x += ov.x; y.y += ov.y; y.z += ov.z; y.w += ov.w;
    }
    *(float4*)&ga.out[ob + i] = y;
  }
}

// ---------------------------------------------------------------------------

#define LAUNCH(Nv, KCv, PREv, EPIv, CATv, GA) \
  gemm_epi_kernel<Nv, KCv, PREv, EPIv, CATv><<<grid, block, 0, stream>>>(GA)

extern "C" void kernel_launch(void* const* d_in, const int* in_sizes, int n_in,
                              void* d_out, int out_size, void* d_ws, size_t ws_size,
                              hipStream_t stream) {
  (void)in_sizes; (void)n_in; (void)out_size; (void)ws_size;

  // Modality order = MODS insertion: market, sentiment, onchain, macro, options
  const int D [5] = {64, 32, 32, 16, 24};
  const int Kp[5] = {64, 32, 32, 32, 32};

  auto F = [&](int i) { return (const float*)d_in[i]; };
  // Leaf order (setup_inputs insertion order):
  //   [0..39]    enc_* : W1,b1,g1,be1,W2,b2,g2,be2   (per modality, 8 each)
  //   [40]       regime (6x256)
  //   [41..400]  ca{l}_{m1}_to_{m2}: Wv,bv,Wo,bo,g,b
  //   [401..406] fusion: W1,b1,g,b,W2,b2
  //   [407..411] x_market..x_options ; [412] regime_idx
  const float* regime = F(40);
  const float* xs[5];
  for (int m = 0; m < 5; ++m) xs[m] = F(407 + m);
  const int* ridx = (const int*)d_in[412];

  const size_t S = (size_t)BROWS * LDIM;     // floats per (B,256) activation
  float* ws = (float*)d_ws;
  float* enc[2][5];
  for (int m = 0; m < 5; ++m) { enc[0][m] = ws + m * S; enc[1][m] = ws + (5 + m) * S; }
  float* tmp   = ws + 10 * S;
  float* xpad3 = ws + 11 * S;                       // padded x_macro  (B x 32)
  float* xpad4 = xpad3 + (size_t)BROWS * 32;        // padded x_options(B x 32)
  __bf16* pool = (__bf16*)(xpad4 + (size_t)BROWS * 32);

  // Packed pool offsets (bf16 elements).
  const size_t poff_enc1[5] = {0, 16384, 24576, 32768, 40960};
  const size_t PW256    = 49152;       // 125 x 65536 : enc W2 (5) then ca (120)
  const size_t POFF_FW1 = PW256 + 125ull * 65536;           // 1280x256
  const size_t POFF_FW2 = POFF_FW1 + 1280ull * 256;         // 256x128

  dim3 block(256);

  // ---------------- preprocessing: pad inputs + pack weights ----------------
  pad_x<<<dim3(BROWS * 32 / 256), block, 0, stream>>>(xs[3], xpad3, 16);
  pad_x<<<dim3(BROWS * 32 / 256), block, 0, stream>>>(xs[4], xpad4, 24);

  for (int m = 0; m < 5; ++m) {
    int nfrag = (Kp[m] >> 5) * 16 * 32;
    pack_w<<<dim3((nfrag + 255) / 256), block, 0, stream>>>(
        F(m * 8 + 0), pool + poff_enc1[m], Kp[m], 256, D[m]);
  }
  {
    PackB pb;
    for (int m = 0; m < 5; ++m) pb.src[m] = F(m * 8 + 4);   // enc W2
    for (int l = 0; l < 3; ++l)
      for (int i = 0; i < 5; ++i) {
        int j = 0;
        for (int m2 = 0; m2 < 5; ++m2) {
          if (m2 == i) continue;
          int cb = 41 + ((l * 5 + i) * 4 + j) * 6;
          int p  = ((l * 5 + i) * 4 + j) * 2;
          pb.src[5 + p + 0] = F(cb + 0);                    // Wv
          pb.src[5 + p + 1] = F(cb + 2);                    // Wo
          ++j;
        }
      }
    pack_w256<<<dim3(16, 125), block, 0, stream>>>(pb, pool + PW256);
  }
  pack_w<<<dim3(40 * 16 * 32 / 256), block, 0, stream>>>(
      F(401), pool + POFF_FW1, 1280, 256, 1280);
  pack_w<<<dim3(8 * 8 * 32 / 256), block, 0, stream>>>(
      F(405), pool + POFF_FW2, 256, 128, 256);

  dim3 grid(BROWS / 16);

  // ---------------- encoders ----------------
  for (int m = 0; m < 5; ++m) {
    int base = m * 8;
    { GArgs ga{};
      ga.A = (m == 3) ? xpad3 : (m == 4) ? xpad4 : xs[m];
      ga.PW = pool + poff_enc1[m];
      ga.bias = F(base + 1); ga.g = F(base + 2); ga.bvec = F(base + 3);
      ga.out = tmp; ga.lda = (m >= 3) ? 32 : D[m];
      if (m == 0) LAUNCH(256, 64, PRE_NONE, EPI_LN, false, ga);
      else        LAUNCH(256, 32, PRE_NONE, EPI_LN, false, ga);
    }
    { GArgs ga{};
      ga.A = tmp; ga.PW = pool + PW256 + (size_t)m * 65536;
      ga.bias = F(base + 5); ga.g = F(base + 6); ga.bvec = F(base + 7);
      ga.regime = regime; ga.ridx = ridx;
      ga.out = enc[0][m]; ga.lda = 256;
      LAUNCH(256, 256, PRE_GELU, EPI_LN_REG, false, ga);
    }
  }

  // ---------------- cross-attention layers ----------------
  int cur = 0;
  for (int l = 0; l < 3; ++l) {
    for (int i = 0; i < 5; ++i) {
      bool first = true;
      int j = 0;
      for (int m2 = 0; m2 < 5; ++m2) {
        if (m2 == i) continue;
        int cb = 41 + ((l * 5 + i) * 4 + j) * 6;
        int p  = ((l * 5 + i) * 4 + j) * 2;
        { GArgs ga{};   // v = enc[m2] @ Wv + bv
          ga.A = enc[cur][m2];
          ga.PW = pool + PW256 + (size_t)(5 + p) * 65536;
          ga.bias = F(cb + 1);
          ga.out = tmp; ga.lda = 256;
          LAUNCH(256, 256, PRE_NONE, EPI_PLAIN, false, ga);
        }
        { GArgs ga{};   // acc (+)= LN(enc[m1] + v@Wo + bo)
          ga.A = tmp;
          ga.PW = pool + PW256 + (size_t)(5 + p + 1) * 65536;
          ga.bias = F(cb + 3); ga.g = F(cb + 4); ga.bvec = F(cb + 5);
          ga.add_in = enc[cur][i]; ga.out = enc[1 - cur][i];
          ga.lda = 256;
          if (first) LAUNCH(256, 256, PRE_NONE, EPI_ATTN_FIRST, false, ga);
          else       LAUNCH(256, 256, PRE_NONE, EPI_ATTN,       false, ga);
        }
        first = false; ++j;
      }
    }
    cur ^= 1;
  }

  // ---------------- fusion head ----------------
  { GArgs ga{};  // tmp = gelu(LN(concat @ W1 + b1)); concat = sorted names
    ga.cat[0] = enc[cur][3];  // macro
    ga.cat[1] = enc[cur][0];  // market
    ga.cat[2] = enc[cur][2];  // onchain
    ga.cat[3] = enc[cur][4];  // options
    ga.cat[4] = enc[cur][1];  // sentiment
    ga.PW = pool + POFF_FW1;
    ga.bias = F(402); ga.g = F(403); ga.bvec = F(404);
    ga.out = tmp;
    LAUNCH(256, 1280, PRE_NONE, EPI_LN_GELU, true, ga);
  }
  { GArgs ga{};  // out = tmp @ W2 + b2  (N = 128)
    ga.A = tmp; ga.PW = pool + POFF_FW2;
    ga.bias = F(406);
    ga.out = (float*)d_out; ga.lda = 256;
    LAUNCH(128, 256, PRE_NONE, EPI_PLAIN, false, ga);
  }
}
